// QINCoInferenceStepEncoderNoSubstep_62775241998555
// MI455X (gfx1250) — matrix operand order repaired and checked by reading, compiled, not run on hardware
//
#include <hip/hip_runtime.h>
#include <hip/hip_bf16.h>

// ---------------------------------------------------------------------------
// QINCo inference step for MI455X (gfx1250, wave32, WMMA bf16 w/ f32 accum,
// TDM tensor_load_to_lds for activation-tile staging)
// Shapes: B=128, F=8, K=256, D=256, H=256, HFF=512, L=2
// ---------------------------------------------------------------------------

typedef __attribute__((ext_vector_type(16))) __bf16   v16bf;
typedef __attribute__((ext_vector_type(8)))  __bf16   v8bf;
typedef __attribute__((ext_vector_type(8)))  float    v8f;
typedef __attribute__((ext_vector_type(4)))  unsigned u32x4;
typedef __attribute__((ext_vector_type(8)))  int      i32x8;
typedef __attribute__((ext_vector_type(4)))  int      i32x4;

#define Bc   128
#define Fc   8
#define Kc   256
#define Dc   256
#define Hc   256
#define HFFc 512
#define BFc  (Bc*Fc)        // 1024
#define NCAND (Fc*Kc)       // 2048

// This toolchain ships the 6-arg TDM builtin (clang-23 / therock-10.0):
//   (uint32x4 g0, int32x8 g1, int32x4 g2, int32x4 g3, int32x8 pad, i32 cpol)
#if defined(__has_builtin)
#  if __has_builtin(__builtin_amdgcn_tensor_load_to_lds) && \
      __has_builtin(__builtin_amdgcn_s_wait_tensorcnt)
#    define QINCO_USE_TDM 1
#  endif
#endif
#ifndef QINCO_USE_TDM
#  define QINCO_USE_TDM 0
#endif

// LDS layout of the fused kernel (byte offsets into dynamic LDS)
#define SOFF_ACT 0                         // 64 x 512 bf16  (64KB)
#define SOFF_HB  65536                     // 64 x 256 bf16  (32KB)
#define SOFF_CF  (65536 + 32768)           // 64 x 256 f32   (64KB)  / P-tile stage
#define SOFF_X   (65536 + 32768 + 65536)   // 256 f32        (1KB)
#define SOFF_Q   (SOFF_X + 1024)           // 512 f32        (2KB)
#define SMEM_SZ  (SOFF_Q + 2048)

// ---- WMMA fragment loaders -------------------------------------------------
// A fragment (16x32 bf16, row-major source, leading dim ld):
//   lane L: m = L%16, hf = L/16
//   VGPR0-3 : K = kb + 16*hf     .. +8   (16 bytes)
//   VGPR4-7 : K = kb + 16*hf + 8 .. +8   (16 bytes)
__device__ inline v16bf load_a_frag(const __bf16* base, int ld, int m, int kb, int hf) {
  union { v16bf v; v8bf p[2]; } u;
  const __bf16* r = base + m * ld + kb + 16 * hf;
  u.p[0] = *(const v8bf*)(r);
  u.p[1] = *(const v8bf*)(r + 8);
  return u.v;
}

// B fragment (32x16 bf16). Source is the TRANSPOSED weight WT[N, Kd]
// (row n = output column, contiguous in K):
//   lane L: n = L%16, hf = L/16
//   VGPR v holds K = {kb + 16*hf + 2v, +1}  => 16 contiguous bf16 = 32 bytes
__device__ inline v16bf load_b_frag(const __bf16* bt, int ld, int n, int kb, int hf) {
  union { v16bf v; v8bf p[2]; } u;
  const __bf16* r = bt + n * ld + kb + 16 * hf;
  u.p[0] = *(const v8bf*)(r);
  u.p[1] = *(const v8bf*)(r + 8);
  return u.v;
}

__device__ inline v8f wmma_bf16(v16bf a, v16bf b, v8f c) {
  return __builtin_amdgcn_wmma_f32_16x16x32_bf16(false, a, false, b,
                                                 (short)0, c, false, false);
}

// ---- tiny conversion kernels ----------------------------------------------
__global__ __launch_bounds__(256) void to_bf16_k(const float* __restrict__ in,
                                                 __bf16* __restrict__ out, int n) {
  int i = blockIdx.x * 256 + threadIdx.x;
  if (i < n) out[i] = (__bf16)in[i];
}

// out[n*Kd + k] = src[(k+rowOff)*N + n]   (WT[N,Kd] from W[Kd,N])
__global__ __launch_bounds__(256) void transpose_bf16_k(const float* __restrict__ src,
                                                        __bf16* __restrict__ out,
                                                        int rowOff, int Kd, int N) {
  int i = blockIdx.x * 256 + threadIdx.x;
  if (i >= Kd * N) return;
  int n = i / Kd, k = i % Kd;
  out[n * Kd + k] = (__bf16)src[(size_t)(k + rowOff) * N + n];
}

// ---- generic WMMA GEMM: C[M,N] = A[M,Kd] @ W  (W given transposed: BT[N,Kd])
__global__ __launch_bounds__(256) void gemm_bf16_k(const __bf16* __restrict__ A,
                                                   const __bf16* __restrict__ BT,
                                                   const float* __restrict__ bias,
                                                   float* __restrict__ Cf,
                                                   __bf16* __restrict__ Cb,
                                                   int M, int N, int Kd) {
  int wave = threadIdx.x >> 5;
  int lane = threadIdx.x & 31;
  int hf   = lane >> 4;
  int ln   = lane & 15;
  int m0   = blockIdx.x * 128 + wave * 16;
  if (m0 >= M) return;

  for (int n0 = 0; n0 < N; n0 += 16) {
    if (n0 + 16 < N)  // pull next weight tile toward the WGP (global_prefetch_b8)
      __builtin_prefetch(BT + (size_t)(n0 + 16 + ln) * Kd, 0, 3);
    v8f acc = {};
    for (int kb = 0; kb < Kd; kb += 32) {
      v16bf a = load_a_frag(A,  Kd, m0 + ln, kb, hf);
      v16bf b = load_b_frag(BT, Kd, n0 + ln, kb, hf);
      acc = wmma_bf16(a, b, acc);
    }
    int n = n0 + ln;
    float bv = bias ? bias[n] : 0.0f;
#pragma unroll
    for (int r = 0; r < 8; ++r) {
      int m = m0 + r + 8 * hf;
      float v = acc[r] + bv;
      if (Cf) Cf[(size_t)m * N + n] = v;
      if (Cb) Cb[(size_t)m * N + n] = (__bf16)v;
    }
  }
}

// ---- fused main kernel -----------------------------------------------------
// grid: (K/64, B*F). block 256 = 8 waves (4 row-waves x 2 col-waves).
// Per (b,f) and 64-candidate tile, keep all activations in LDS through the
// whole residual MLP + W_out + distance epilogue. Emits dist' = |c|^2 - 2 x.c
__global__ __launch_bounds__(256) void qinco_main_k(
    const float*  __restrict__ P,     // [K, HFF]   A@W1[0]      (f32)
    const __bf16* __restrict__ Pb,    // [K, HFF]   same, bf16 (TDM source)
    const float*  __restrict__ Q,     // [BF, HFF]  Bx@W1[0]+b1[0]
    const float*  __restrict__ Af,    // [K, H]     h0@Wc
    const float*  __restrict__ Bxf,   // [BF, H]    xhat@Wx+b_cat
    const __bf16* __restrict__ W20T,  // [H, HFF]
    const __bf16* __restrict__ W11T,  // [HFF, H]
    const __bf16* __restrict__ W21T,  // [H, HFF]
    const __bf16* __restrict__ WoutT, // [D, H]
    const float* __restrict__ b2,     // [L, H]
    const float* __restrict__ b1,     // [L, HFF]
    const float* __restrict__ bout,   // [D]
    const float* __restrict__ cb,     // [K, D]
    const float* __restrict__ xhat,   // [BF, D]
    const float* __restrict__ xB,     // [B, D]
    float* __restrict__ dists)        // [B, F*K]
{
  extern __shared__ char smem[];
  __bf16* s_act = (__bf16*)(smem + SOFF_ACT);  // 64 x 512 bf16
  __bf16* s_hb  = (__bf16*)(smem + SOFF_HB);   // 64 x 256 bf16
  float*  s_cf  = (float*)(smem + SOFF_CF);    // 64 x 256 f32 (h1 / c)
  __bf16* s_pb  = (__bf16*)(smem + SOFF_CF);   // aliases s_cf during staging
  float*  s_x   = (float*)(smem + SOFF_X);     // 256 f32
  float*  s_q   = (float*)(smem + SOFF_Q);     // 512 f32

  const int bf = blockIdx.y;
  const int b  = bf >> 3;
  const int f  = bf & 7;
  const int k0 = blockIdx.x * 64;
  const int tid  = threadIdx.x;
  const int wave = tid >> 5, lane = tid & 31;
  const int hf = lane >> 4, ln = lane & 15;
  const int mw = (wave & 3) * 16;   // row offset of this wave in the 64-tile
  const int cw = wave >> 2;         // column half (0/1)

  s_x[tid] = xB[b * Dc + tid];
  s_q[tid]       = Q[(size_t)bf * HFFc + tid];
  s_q[tid + 256] = Q[(size_t)bf * HFFc + tid + 256];

#if QINCO_USE_TDM
  // --- TDM: DMA the 64x512 bf16 P-tile into LDS (one descriptor, wave 0) ---
  if (wave == 0) {
    unsigned long long ga =
        (unsigned long long)(const void*)(Pb + (size_t)k0 * HFFc);
    u32x4 g0;
    g0.x = 1u;                                   // count=1, user mode, no gather
    g0.y = (unsigned)SOFF_CF;                    // lds_addr (bytes)
    g0.z = (unsigned)(ga & 0xffffffffu);         // global_addr[31:0]
    g0.w = (unsigned)((ga >> 32) & 0x01ffffffu)  // global_addr[56:32]
         | (2u << 30);                           // type = 2 ("image")
    i32x8 g1;
    g1[0] = (1 << 16);                 // workgroup_mask=0, data_size=1 (2B)
    g1[1] = (HFFc & 0xffff) << 16;     // tensor_dim0[15:0]  = 512
    g1[2] = (Kc & 0xffff) << 16;       // tensor_dim0 hi=0 | tensor_dim1[15:0]=256
    g1[3] = (HFFc & 0xffff) << 16;     // tensor_dim1 hi=0 | tile_dim0 = 512
    g1[4] = 64;                        // tile_dim1 = 64, tile_dim2 = 0
    g1[5] = HFFc;                      // tensor_dim0_stride[31:0] = 512
    g1[6] = 0;                         // stride hi, tensor_dim1_stride lo
    g1[7] = 0;
    i32x4 gz4 = {0, 0, 0, 0};          // groups 2/3 unused (2-D tile)
    i32x8 gz8 = {0, 0, 0, 0, 0, 0, 0, 0};
    __builtin_amdgcn_tensor_load_to_lds(g0, g1, gz4, gz4, gz8, 0);
    __builtin_amdgcn_s_wait_tensorcnt(0);   // TENSORcnt is per-wave
  }
  __syncthreads();                          // publish LDS tile to all waves

  // z0 = relu(P[k0+m] + Q[bf])  -> s_act bf16 [64 x 512]
  for (int e = tid; e < 64 * HFFc; e += 256) {
    int j = e & 511;
    float v = (float)s_pb[e] + s_q[j];
    s_act[e] = (__bf16)fmaxf(v, 0.0f);
  }
#else
  __syncthreads();
  for (int e = tid; e < 64 * HFFc; e += 256) {
    int m = e >> 9, j = e & 511;
    float v = P[(size_t)(k0 + m) * HFFc + j] + s_q[j];
    s_act[e] = (__bf16)fmaxf(v, 0.0f);
  }
#endif
  __syncthreads();

  // GEMM1: h1 = z0 @ W2[0] + b2[0] + A[k] + Bx[bf]   (64x512 @ 512x256)
  {
    const float* b2_0 = b2;
    for (int t = 0; t < 8; ++t) {
      int n0 = cw * 128 + t * 16;
      v8f acc = {};
      for (int kb = 0; kb < HFFc; kb += 32) {
        v16bf a = load_a_frag(s_act, HFFc, mw + ln, kb, hf);
        v16bf w = load_b_frag(W20T,  HFFc, n0 + ln, kb, hf);
        acc = wmma_bf16(a, w, acc);
      }
      int n = n0 + ln;
#pragma unroll
      for (int r = 0; r < 8; ++r) {
        int m = mw + r + 8 * hf;
        float v = acc[r] + b2_0[n] + Af[(size_t)(k0 + m) * Hc + n]
                         + Bxf[(size_t)bf * Hc + n];
        s_cf[m * Hc + n] = v;          // keep h1 in f32 for the residual
        s_hb[m * Hc + n] = (__bf16)v;  // bf16 operand for next GEMM
      }
    }
  }
  __syncthreads();

  // GEMM2: z1 = relu(h1 @ W1[1] + b1[1])   (64x256 @ 256x512) -> s_act
  {
    const float* b1_1 = b1 + HFFc;
    for (int t = 0; t < 16; ++t) {
      int n0 = cw * 256 + t * 16;
      v8f acc = {};
      for (int kb = 0; kb < Hc; kb += 32) {
        v16bf a = load_a_frag(s_hb, Hc, mw + ln, kb, hf);
        v16bf w = load_b_frag(W11T, Hc, n0 + ln, kb, hf);
        acc = wmma_bf16(a, w, acc);
      }
      int n = n0 + ln;
#pragma unroll
      for (int r = 0; r < 8; ++r) {
        int m = mw + r + 8 * hf;
        s_act[m * HFFc + n] = (__bf16)fmaxf(acc[r] + b1_1[n], 0.0f);
      }
    }
  }
  __syncthreads();

  // GEMM3: h2 = h1 + z1 @ W2[1] + b2[1]   (64x512 @ 512x256) -> s_hb
  {
    const float* b2_1 = b2 + Hc;
    for (int t = 0; t < 8; ++t) {
      int n0 = cw * 128 + t * 16;
      v8f acc = {};
      for (int kb = 0; kb < HFFc; kb += 32) {
        v16bf a = load_a_frag(s_act, HFFc, mw + ln, kb, hf);
        v16bf w = load_b_frag(W21T,  HFFc, n0 + ln, kb, hf);
        acc = wmma_bf16(a, w, acc);
      }
      int n = n0 + ln;
#pragma unroll
      for (int r = 0; r < 8; ++r) {
        int m = mw + r + 8 * hf;
        float v = s_cf[m * Hc + n] + acc[r] + b2_1[n];
        s_hb[m * Hc + n] = (__bf16)v;
      }
    }
  }
  __syncthreads();

  // GEMM4: c = h2 @ W_out + b_out + cb[k] + xhat[bf]   (64x256 @ 256x256)
  {
    for (int t = 0; t < 8; ++t) {
      int n0 = cw * 128 + t * 16;
      v8f acc = {};
      for (int kb = 0; kb < Hc; kb += 32) {
        v16bf a = load_a_frag(s_hb,  Hc, mw + ln, kb, hf);
        v16bf w = load_b_frag(WoutT, Hc, n0 + ln, kb, hf);
        acc = wmma_bf16(a, w, acc);
      }
      int n = n0 + ln;
#pragma unroll
      for (int r = 0; r < 8; ++r) {
        int m = mw + r + 8 * hf;
        float v = acc[r] + bout[n] + cb[(size_t)(k0 + m) * Dc + n]
                         + xhat[(size_t)bf * Dc + n];
        s_cf[m * Dc + n] = v;
      }
    }
  }
  __syncthreads();

  // distance epilogue: dist' = |c|^2 - 2 x.c  (||x||^2 constant per b, dropped)
  if (tid < 64) {
    int m = tid;
    float ss = 0.0f, dx = 0.0f;
    for (int d = 0; d < Dc; ++d) {
      float c = s_cf[m * Dc + d];
      ss += c * c;
      dx += c * s_x[d];
    }
    dists[(size_t)b * NCAND + f * Kc + (k0 + m)] = ss - 2.0f * dx;
  }
}

// ---- per-b argmin + f32 recompute of the winning candidate -----------------
__global__ __launch_bounds__(256) void argmin_recompute_k(
    const float* __restrict__ dists,
    const float* __restrict__ P,  const float* __restrict__ Q,
    const float* __restrict__ Af, const float* __restrict__ Bxf,
    const float* __restrict__ W2, const float* __restrict__ W1,
    const float* __restrict__ Wout,
    const float* __restrict__ b1v, const float* __restrict__ b2v,
    const float* __restrict__ bout,
    const float* __restrict__ cb, const float* __restrict__ xhat,
    const long long* __restrict__ codes,
    float* __restrict__ out)
{
  __shared__ float bestd[256];
  __shared__ int   besti[256];
  __shared__ float z0f[HFFc], h1f[Hc], z1f[HFFc], h2f[Hc];
  __shared__ int   s_idx;

  const int b = blockIdx.x, t = threadIdx.x;

  // argmin with first-index tie-break (matches jnp.argmin)
  float bd = 3.402823466e38f; int bi = 0;
  for (int i = t; i < NCAND; i += 256) {
    float d = dists[(size_t)b * NCAND + i];
    if (d < bd) { bd = d; bi = i; }
  }
  bestd[t] = bd; besti[t] = bi;
  __syncthreads();
  if (t == 0) {
    float m = bestd[0]; int mi = besti[0];
    for (int i = 1; i < 256; ++i) {
      if (bestd[i] < m || (bestd[i] == m && besti[i] < mi)) { m = bestd[i]; mi = besti[i]; }
    }
    s_idx = mi;
  }
  __syncthreads();

  const int idx = s_idx;
  const int f = idx >> 8, k = idx & 255;
  const int bfi = b * Fc + f;

  for (int j = t; j < HFFc; j += 256)
    z0f[j] = fmaxf(P[(size_t)k * HFFc + j] + Q[(size_t)bfi * HFFc + j], 0.0f);
  __syncthreads();

  { // h1 = A[k] + Bx[bf] + b2[0] + z0 @ W2[0]
    float s = Af[(size_t)k * Hc + t] + Bxf[(size_t)bfi * Hc + t] + b2v[t];
    for (int j = 0; j < HFFc; ++j) s += z0f[j] * W2[(size_t)j * Hc + t];
    h1f[t] = s;
  }
  __syncthreads();

  for (int j = t; j < HFFc; j += 256) { // z1 = relu(h1 @ W1[1] + b1[1])
    float s = b1v[HFFc + j];
    const float* W1_1 = W1 + (size_t)Hc * HFFc;
    for (int i = 0; i < Hc; ++i) s += h1f[i] * W1_1[(size_t)i * HFFc + j];
    z1f[j] = fmaxf(s, 0.0f);
  }
  __syncthreads();

  { // h2 = h1 + z1 @ W2[1] + b2[1]
    float s = h1f[t] + b2v[Hc + t];
    const float* W2_1 = W2 + (size_t)HFFc * Hc;
    for (int j = 0; j < HFFc; ++j) s += z1f[j] * W2_1[(size_t)j * Hc + t];
    h2f[t] = s;
  }
  __syncthreads();

  { // c = h2 @ W_out + b_out + cb[k] + xhat[bf]  -> xhat_next[b, 0, :]
    float s = bout[t] + cb[(size_t)k * Dc + t] + xhat[(size_t)bfi * Dc + t];
    for (int i = 0; i < Hc; ++i) s += h2f[i] * Wout[(size_t)i * Dc + t];
    out[(size_t)b * Dc + t] = s;
  }

  if (t == 0) { // codes_new = concat(codes_MBF, idx)  as int64 after the floats
    long long* co = (long long*)(out + (size_t)Bc * Dc);
    co[b]           = codes[b];
    co[Bc + b]      = codes[Bc + b];
    co[2 * Bc + b]  = (long long)idx;
  }
}

// ---------------------------------------------------------------------------
extern "C" void kernel_launch(void* const* d_in, const int* in_sizes, int n_in,
                              void* d_out, int out_size, void* d_ws, size_t ws_size,
                              hipStream_t stream) {
  (void)in_sizes; (void)n_in; (void)out_size; (void)ws_size;

  const float*     x     = (const float*)d_in[0];
  const float*     xhat  = (const float*)d_in[1];
  const long long* codes = (const long long*)d_in[2];
  const float*     cb    = (const float*)d_in[3];
  const float*     W_in  = (const float*)d_in[4];
  const float*     b_in  = (const float*)d_in[5];
  const float*     W_cat = (const float*)d_in[6];
  const float*     b_cat = (const float*)d_in[7];
  const float*     W1    = (const float*)d_in[8];
  const float*     b1    = (const float*)d_in[9];
  const float*     W2    = (const float*)d_in[10];
  const float*     b2    = (const float*)d_in[11];
  const float*     Wout  = (const float*)d_in[12];
  const float*     bout  = (const float*)d_in[13];
  float* out = (float*)d_out;

  // workspace carving (256B aligned regions), ~8 MB total
  size_t o = 0;
  char* wsb = (char*)d_ws;
  auto carve = [&](size_t bytes) -> char* {
    char* p = wsb + o; o += (bytes + 255) & ~(size_t)255; return p;
  };
  __bf16* cb_bf   = (__bf16*)carve((size_t)Kc * Dc * 2);
  __bf16* xhat_bf = (__bf16*)carve((size_t)BFc * Dc * 2);
  __bf16* WinT    = (__bf16*)carve((size_t)Hc * Dc * 2);
  __bf16* WcT     = (__bf16*)carve((size_t)Hc * Hc * 2);
  __bf16* WxT     = (__bf16*)carve((size_t)Hc * Dc * 2);
  __bf16* W10T    = (__bf16*)carve((size_t)HFFc * Hc * 2);
  __bf16* W20T    = (__bf16*)carve((size_t)Hc * HFFc * 2);
  __bf16* W11T    = (__bf16*)carve((size_t)HFFc * Hc * 2);
  __bf16* W21T    = (__bf16*)carve((size_t)Hc * HFFc * 2);
  __bf16* WoutT   = (__bf16*)carve((size_t)Dc * Hc * 2);
  __bf16* h0_bf   = (__bf16*)carve((size_t)Kc * Hc * 2);
  float*  Af      = (float*) carve((size_t)Kc * Hc * 4);
  __bf16* A_bf    = (__bf16*)carve((size_t)Kc * Hc * 2);
  float*  Pf      = (float*) carve((size_t)Kc * HFFc * 4);
  __bf16* Pb      = (__bf16*)carve((size_t)Kc * HFFc * 2);
  float*  Bxf     = (float*) carve((size_t)BFc * Hc * 4);
  __bf16* Bx_bf   = (__bf16*)carve((size_t)BFc * Hc * 2);
  float*  Qf      = (float*) carve((size_t)BFc * HFFc * 4);
  float*  dists   = (float*) carve((size_t)Bc * NCAND * 4);

  // 1) bf16 conversions + weight transposes (all tiny, L2-resident after)
  to_bf16_k<<<(Kc*Dc + 255)/256, 256, 0, stream>>>(cb, cb_bf, Kc*Dc);
  to_bf16_k<<<(BFc*Dc + 255)/256, 256, 0, stream>>>(xhat, xhat_bf, BFc*Dc);
  transpose_bf16_k<<<(Dc*Hc + 255)/256, 256, 0, stream>>>(W_in,  WinT,  0,   Dc, Hc);
  transpose_bf16_k<<<(Hc*Hc + 255)/256, 256, 0, stream>>>(W_cat, WcT,   0,   Hc, Hc);
  transpose_bf16_k<<<(Dc*Hc + 255)/256, 256, 0, stream>>>(W_cat, WxT,   Hc,  Dc, Hc);
  transpose_bf16_k<<<(Hc*HFFc + 255)/256, 256, 0, stream>>>(W1,           W10T, 0, Hc,   HFFc);
  transpose_bf16_k<<<(HFFc*Hc + 255)/256, 256, 0, stream>>>(W2,           W20T, 0, HFFc, Hc);
  transpose_bf16_k<<<(Hc*HFFc + 255)/256, 256, 0, stream>>>(W1 + Hc*HFFc, W11T, 0, Hc,   HFFc);
  transpose_bf16_k<<<(HFFc*Hc + 255)/256, 256, 0, stream>>>(W2 + HFFc*Hc, W21T, 0, HFFc, Hc);
  transpose_bf16_k<<<(Hc*Dc + 255)/256, 256, 0, stream>>>(Wout,  WoutT, 0,   Hc, Dc);

  // 2) precompute GEMMs (WMMA)
  // h0 = cb @ W_in + b_in
  gemm_bf16_k<<<Kc/128, 256, 0, stream>>>(cb_bf, WinT, b_in, nullptr, h0_bf, Kc, Hc, Dc);
  // A = h0 @ Wc
  gemm_bf16_k<<<Kc/128, 256, 0, stream>>>(h0_bf, WcT, nullptr, Af, A_bf, Kc, Hc, Hc);
  // P = A @ W1[0]   (f32 for recompute, bf16 for the TDM-staged fused kernel)
  gemm_bf16_k<<<Kc/128, 256, 0, stream>>>(A_bf, W10T, nullptr, Pf, Pb, Kc, HFFc, Hc);
  // Bx = xhat @ Wx + b_cat
  gemm_bf16_k<<<BFc/128, 256, 0, stream>>>(xhat_bf, WxT, b_cat, Bxf, Bx_bf, BFc, Hc, Dc);
  // Q = Bx @ W1[0] + b1[0]
  gemm_bf16_k<<<BFc/128, 256, 0, stream>>>(Bx_bf, W10T, b1, Qf, nullptr, BFc, HFFc, Hc);

  // 3) fused MLP + distance kernel
  qinco_main_k<<<dim3(Kc/64, BFc), 256, SMEM_SZ, stream>>>(
      Pf, Pb, Qf, Af, Bxf, W20T, W11T, W21T, WoutT,
      b2, b1, bout, cb, xhat, x, dists);

  // 4) per-b argmin + f32 recompute of winner + codes concat
  argmin_recompute_k<<<Bc, 256, 0, stream>>>(
      dists, Pf, Qf, Af, Bxf, W2, W1, Wout, b1, b2, bout, cb, xhat, codes, out);
}
